// HybridGemmaDiT_6614249636444
// MI455X (gfx1250) — matrix operand order repaired and checked by reading, not compiled, not run on hardware
//
#include <hip/hip_runtime.h>
#include <hip/hip_bf16.h>
#include <math.h>

// ---------------------------------------------------------------------------
// MoE (top-3 of 8 experts, SwiGLU) for MI455X / gfx1250, bf16 WMMA path.
// v2: weights pre-transposed + pre-converted to bf16 in workspace so the
//     WMMA B-operands are fed by 2x global_load_b128 per fragment (K-contiguous
//     per lane) with no fp32->bf16 cvt in the hot loop.
// ---------------------------------------------------------------------------

#define T_TOK 4096          // 2 * 2048 tokens
#define DIMC  1024
#define HIDC  4096
#define NEXP  8
#define TOPK  3
#define BM    64            // tokens per ffn block
#define HCHUNK 128          // hidden columns processed per fused iteration
#define XS    (DIMC + 8)    // bf16 row stride of x tile in LDS
#define HS    (HCHUNK + 8)  // bf16 row stride of hidden chunk in LDS

typedef __attribute__((ext_vector_type(16))) __bf16 v16bf;
typedef __attribute__((ext_vector_type(8)))  __bf16 v8bf;
typedef __attribute__((ext_vector_type(8)))  float  v8f;

static __device__ inline v8f wmma_bf16(v16bf a, v16bf b, v8f c) {
  // v_wmma_f32_16x16x32_bf16: (neg_a, A, neg_b, B, c_mod, C, reuse_a, reuse_b)
  return __builtin_amdgcn_wmma_f32_16x16x32_bf16(false, a, false, b, (short)0, c,
                                                 false, false);
}

// A-fragment (16x32 bf16, M x K) from LDS, row-major, stride `rs` (bf16 units,
// multiple of 8 so 16B runs stay aligned).
// ISA layout: lanes 0-15 -> M=lane, K = {0..7, 16..23}; lanes 16-31 -> M=lane-16,
// K = {8..15, 24..31}.  Two aligned 16B LDS vector loads per lane.
static __device__ inline v16bf lds_afrag(const __bf16* p, int rs, int row0, int kbase) {
  const int lane = threadIdx.x & 31;
  const int m    = lane & 15;
  const int k0   = kbase + ((lane >> 4) << 3);   // +8 for upper half-wave
  const __bf16* r = p + (row0 + m) * rs + k0;
  const v8bf lo = *(const v8bf*)r;         // K = k0..k0+7
  const v8bf hi = *(const v8bf*)(r + 16);  // K = k0+16..k0+23
  v16bf a;
#pragma unroll
  for (int i = 0; i < 8; ++i) { a[i] = lo[i]; a[8 + i] = hi[i]; }
  return a;
}

// B-fragment (32x16 bf16, K x N) from pre-transposed bf16 weights wt[n][k]
// (N-major, ldk = K length). ISA layout: lanes 0-15 -> N=lane, K={0..7,16..23};
// lanes 16-31 -> N=lane-16, K={8..15,24..31} (relative to kbase).
// two 8-bf16 runs per lane (k0.., k0+16..) = two aligned global_load_b128.
static __device__ inline v16bf ldg_bfrag(const __bf16* __restrict__ wt, int ldk,
                                         int kbase, int ncol0) {
  const int lane = threadIdx.x & 31;
  const int n    = ncol0 + (lane & 15);
  const int k0   = kbase + ((lane >> 4) << 3);   // +8 for upper half-wave
  const __bf16* p = wt + (size_t)n * (size_t)ldk + k0;
  const v8bf lo = *(const v8bf*)p;
  const v8bf hi = *(const v8bf*)(p + 16);
  v16bf b;
#pragma unroll
  for (int i = 0; i < 8; ++i) { b[i] = lo[i]; b[8 + i] = hi[i]; }
  return b;
}

// ---------------------------------------------------------------------------
// 0) Weight prep: transpose [R][C] fp32 row-major -> [C][R] bf16 row-major,
//    per expert (blockIdx.z). Tiled 32x32 through LDS, coalesced both sides.
// ---------------------------------------------------------------------------
__global__ __launch_bounds__(256)
void transpose_cvt_kernel(const float* __restrict__ src, __bf16* __restrict__ dst,
                          int R, int C, size_t stride_e) {
  const int e = blockIdx.z;
  src += (size_t)e * stride_e;
  dst += (size_t)e * stride_e;
  __shared__ float tile[32][33];
  const int c0 = blockIdx.x * 32;             // input col block == output row block
  const int r0 = blockIdx.y * 32;             // input row block
  const int tx = threadIdx.x & 31;
  const int ty = threadIdx.x >> 5;            // 0..7
#pragma unroll
  for (int i = 0; i < 4; ++i) {
    const int r = r0 + ty + i * 8;
    tile[ty + i * 8][tx] = src[(size_t)r * C + (c0 + tx)];
  }
  __syncthreads();
#pragma unroll
  for (int i = 0; i < 4; ++i) {
    const int c = c0 + ty + i * 8;            // output row
    dst[(size_t)c * R + (r0 + tx)] = (__bf16)tile[tx][ty + i * 8];
  }
}

// ---------------------------------------------------------------------------
// 1) Router: logits, sigmoid, top-3, dense combine weights, aux partials.
// ---------------------------------------------------------------------------
__global__ void router_kernel(const float* __restrict__ x,
                              const float* __restrict__ rw,
                              const float* __restrict__ rb,
                              float* __restrict__ dense_w,
                              float* __restrict__ aux_partial) {
  const int tok  = blockIdx.x;
  const int wv   = threadIdx.x >> 5;   // 8 waves -> 8 experts
  const int lane = threadIdx.x & 31;
  __shared__ float logits_s[NEXP];

  const float* xr = x  + (size_t)tok * DIMC;
  const float* wr = rw + (size_t)wv  * DIMC;
  float s = 0.f;
  for (int d = lane; d < DIMC; d += 32) s += xr[d] * wr[d];
#pragma unroll
  for (int off = 16; off > 0; off >>= 1) s += __shfl_down(s, off, 32);
  if (lane == 0) logits_s[wv] = s + rb[wv];
  __syncthreads();

  if (threadIdx.x == 0) {
    float lg[NEXP], sc[NEXP];
    float sq = 0.f;
#pragma unroll
    for (int e = 0; e < NEXP; ++e) {
      lg[e] = logits_s[e];
      sq   += lg[e] * lg[e];
      sc[e] = 1.f / (1.f + __expf(-lg[e]));
    }
    aux_partial[tok] = sq;

    int   sel[TOPK];
    float sv[TOPK];
    bool  used[NEXP] = {};
    float tsum = 0.f;
#pragma unroll
    for (int k = 0; k < TOPK; ++k) {
      int best = 0; float bv = -1e30f;
#pragma unroll
      for (int e = 0; e < NEXP; ++e)
        if (!used[e] && sc[e] > bv) { bv = sc[e]; best = e; }  // jax tie-break: first
      used[best] = true; sel[k] = best; sv[k] = bv; tsum += bv;
    }
    const float inv = 1.f / (tsum + 1e-6f);
    float dw[NEXP] = {};
#pragma unroll
    for (int k = 0; k < TOPK; ++k) dw[sel[k]] = sv[k] * inv;
#pragma unroll
    for (int e = 0; e < NEXP; ++e) dense_w[(size_t)tok * NEXP + e] = dw[e];
  }
}

// ---------------------------------------------------------------------------
// 2) Deterministic per-expert compaction (one wave32 per expert, ballot scan).
// ---------------------------------------------------------------------------
__global__ void compact_kernel(const float* __restrict__ dense_w,
                               int* __restrict__ counts,
                               int* __restrict__ idx,
                               float* __restrict__ wgt) {
  const int e    = blockIdx.x;
  const int lane = threadIdx.x;   // 32 threads
  int base = 0;
  for (int c = 0; c < T_TOK / 32; ++c) {
    const int tok = c * 32 + lane;
    const float wv = dense_w[(size_t)tok * NEXP + e];
    const unsigned mask = (unsigned)__ballot(wv > 0.f);
    const int pre = __builtin_popcount(mask & ((1u << lane) - 1u));
    if (wv > 0.f) {
      idx[e * T_TOK + base + pre] = tok;
      wgt[e * T_TOK + base + pre] = wv;
    }
    base += __builtin_popcount(mask);
  }
  if (lane == 0) counts[e] = base;
}

// ---------------------------------------------------------------------------
// 3) aux_loss = 0.01 * mean(logits^2), fixed-order reduction (deterministic).
// ---------------------------------------------------------------------------
__global__ void aux_kernel(const float* __restrict__ aux_partial,
                           float* __restrict__ out_aux) {
  __shared__ float red[256];
  float s = 0.f;
  for (int i = threadIdx.x; i < T_TOK; i += 256) s += aux_partial[i];
  red[threadIdx.x] = s;
  __syncthreads();
  for (int o = 128; o > 0; o >>= 1) {
    if ((int)threadIdx.x < o) red[threadIdx.x] += red[threadIdx.x + o];
    __syncthreads();
  }
  if (threadIdx.x == 0) *out_aux = 0.01f * red[0] / (float)(T_TOK * NEXP);
}

// ---------------------------------------------------------------------------
// 4) Fused expert FFN: per block = (expert, 64 gathered tokens).
//    Loop over 32 hidden chunks of 128:
//      Phase A: hidden = silu(x@w1) * (x@w2) for the chunk   (bf16 WMMA)
//      Phase B: out_acc += hidden @ w3[chunk rows]           (bf16 WMMA)
//    Epilogue: scale by router weight, atomicAdd into out.
// ---------------------------------------------------------------------------
extern "C" __global__ __launch_bounds__(256)
void moe_ffn_kernel(const float* __restrict__ x,
                    const __bf16* __restrict__ w12t,   // [E][2*HID][DIM] bf16
                    const __bf16* __restrict__ w3t,    // [E][DIM][HID]  bf16
                    const int*   __restrict__ counts,
                    const int*   __restrict__ idx,
                    const float* __restrict__ wgt,
                    float* __restrict__ out) {
  const int e     = blockIdx.y;
  const int cnt   = counts[e];
  const int start = blockIdx.x * BM;
  if (start >= cnt) return;

  extern __shared__ char smem_raw[];
  __bf16* x_lds  = (__bf16*)smem_raw;                                   // BM*XS bf16
  __bf16* h_lds  = (__bf16*)(smem_raw + (size_t)BM * XS * 2);           // BM*HS bf16
  int*    tok_s  = (int*)  (smem_raw + (size_t)BM * XS * 2 + (size_t)BM * HS * 2);
  float*  wgt_s  = (float*)(tok_s + BM);

  if (threadIdx.x < BM) {
    const int r = start + (int)threadIdx.x;
    if (r < cnt) {
      tok_s[threadIdx.x] = idx[e * T_TOK + r];
      wgt_s[threadIdx.x] = wgt[e * T_TOK + r];
    } else {
      tok_s[threadIdx.x] = 0;     // valid address; weight 0 kills contribution
      wgt_s[threadIdx.x] = 0.f;
    }
  }
  __syncthreads();

  // Stage x tile (fp32 -> bf16), coalesced along dim.
  for (int i = threadIdx.x; i < BM * DIMC; i += 256) {
    const int r = i >> 10;             // / DIMC
    const int d = i & (DIMC - 1);
    x_lds[r * XS + d] = (__bf16)x[(size_t)tok_s[r] * DIMC + d];
  }
  __syncthreads();

  const int wv   = threadIdx.x >> 5;   // wave id 0..7
  const int lane = threadIdx.x & 31;
  const __bf16* w12e = w12t + (size_t)e * (2 * HIDC) * DIMC;  // rows: n of w1|w2, len DIMC
  const __bf16* w3e  = w3t  + (size_t)e * DIMC * HIDC;        // rows: n of out, len HIDC

  v8f acc[4][8];                       // 4 M-tiles x 8 N-tiles of out (per wave: 128 cols)
#pragma unroll
  for (int mt = 0; mt < 4; ++mt)
#pragma unroll
    for (int nt = 0; nt < 8; ++nt) acc[mt][nt] = (v8f){};

  for (int hc = 0; hc < HIDC / HCHUNK; ++hc) {
    // ---------------- Phase A: SwiGLU hidden chunk -----------------------
    v8f c1[4] = {}, c2[4] = {};
    const int col1 = hc * HCHUNK + wv * 16;      // this wave's 16 cols of w1
#pragma unroll 4
    for (int ks = 0; ks < DIMC / 32; ++ks) {
      const int kb = ks * 32;
      const v16bf b1 = ldg_bfrag(w12e, DIMC, kb, col1);         // w1 rows
      const v16bf b2 = ldg_bfrag(w12e, DIMC, kb, HIDC + col1);  // w2 rows
#pragma unroll
      for (int mt = 0; mt < 4; ++mt) {
        const v16bf a = lds_afrag(x_lds, XS, mt * 16, kb);
        c1[mt] = wmma_bf16(a, b1, c1[mt]);
        c2[mt] = wmma_bf16(a, b2, c2[mt]);
      }
    }
    // silu(x1)*x2 -> h_lds (C layout: VGPR i, lane l: M = i + 8*(l>=16), N = l&15)
    {
      const int half = lane >> 4;
      const int coln = wv * 16 + (lane & 15);
#pragma unroll
      for (int mt = 0; mt < 4; ++mt)
#pragma unroll
        for (int i = 0; i < 8; ++i) {
          const int row = mt * 16 + i + half * 8;
          const float a1 = c1[mt][i];
          const float a2 = c2[mt][i];
          const float hv = (a1 / (1.f + __expf(-a1))) * a2;
          h_lds[row * HS + coln] = (__bf16)hv;
        }
    }
    __syncthreads();

    // ---------------- Phase B: out_acc += hidden @ w3 --------------------
#pragma unroll
    for (int ks = 0; ks < HCHUNK / 32; ++ks) {
      const int kb = ks * 32;
      v16bf af[4];
#pragma unroll
      for (int mt = 0; mt < 4; ++mt) af[mt] = lds_afrag(h_lds, HS, mt * 16, kb);
#pragma unroll
      for (int nt = 0; nt < 8; ++nt) {
        const v16bf b = ldg_bfrag(w3e, HIDC, hc * HCHUNK + kb, wv * 128 + nt * 16);
#pragma unroll
        for (int mt = 0; mt < 4; ++mt) acc[mt][nt] = wmma_bf16(af[mt], b, acc[mt][nt]);
      }
    }
    __syncthreads();   // before next iteration overwrites h_lds
  }

  // ---------------- Epilogue: scale by router weight, atomic combine ------
  {
    const int half = lane >> 4;
    const int nn   = lane & 15;
#pragma unroll
    for (int mt = 0; mt < 4; ++mt)
#pragma unroll
      for (int i = 0; i < 8; ++i) {
        const int row = mt * 16 + i + half * 8;
        const float wt = wgt_s[row];
        if (wt != 0.f) {
          const size_t obase = (size_t)tok_s[row] * DIMC;
#pragma unroll
          for (int nt = 0; nt < 8; ++nt) {
            const int col = wv * 128 + nt * 16 + nn;
            atomicAdd(out + obase + col, acc[mt][nt][i] * wt);
          }
        }
      }
  }
}

// ---------------------------------------------------------------------------
// Host launcher
// ---------------------------------------------------------------------------
extern "C" void kernel_launch(void* const* d_in, const int* in_sizes, int n_in,
                              void* d_out, int out_size, void* d_ws, size_t ws_size,
                              hipStream_t stream) {
  const float* x   = (const float*)d_in[0];
  const float* rw  = (const float*)d_in[1];
  const float* rb  = (const float*)d_in[2];
  const float* w12 = (const float*)d_in[3];
  const float* w3  = (const float*)d_in[4];
  float* out = (float*)d_out;

  // Workspace layout (every region fully rewritten each launch -> deterministic)
  char* ws = (char*)d_ws;
  const size_t W12T_BYTES = (size_t)NEXP * 2 * HIDC * DIMC * 2;   // 128 MB bf16
  const size_t W3T_BYTES  = (size_t)NEXP * DIMC * HIDC * 2;       //  64 MB bf16
  __bf16* w12t = (__bf16*)ws;
  __bf16* w3t  = (__bf16*)(ws + W12T_BYTES);
  char*  rws   = ws + W12T_BYTES + W3T_BYTES;
  float* dense_w     = (float*)rws;                          // T*E f32   (128 KB)
  float* aux_partial = (float*)(rws + 131072);               // T f32     ( 16 KB)
  int*   counts      = (int*)  (rws + 147456);               // E i32
  int*   idxb        = (int*)  (rws + 147520);               // E*T i32   (128 KB)
  float* wgtb        = (float*)(rws + 147520 + 131072);      // E*T f32   (128 KB)

  // Zero the dense output region (atomic accumulation target).
  hipMemsetAsync(d_out, 0, (size_t)T_TOK * DIMC * sizeof(float), stream);

  // Weight prep: fp32 [R][C] -> bf16 [C][R] per expert (one-shot, ~25us @23.3TB/s).
  transpose_cvt_kernel<<<dim3((2 * HIDC) / 32, DIMC / 32, NEXP), 256, 0, stream>>>(
      w12, w12t, DIMC, 2 * HIDC, (size_t)DIMC * 2 * HIDC);
  transpose_cvt_kernel<<<dim3(DIMC / 32, HIDC / 32, NEXP), 256, 0, stream>>>(
      w3, w3t, HIDC, DIMC, (size_t)HIDC * DIMC);

  router_kernel <<<T_TOK, 256, 0, stream>>>(x, rw, rb, dense_w, aux_partial);
  compact_kernel<<<NEXP, 32, 0, stream>>>(dense_w, counts, idxb, wgtb);
  aux_kernel    <<<1, 256, 0, stream>>>(aux_partial, out + (size_t)T_TOK * DIMC);

  const size_t smem = (size_t)BM * XS * 2 + (size_t)BM * HS * 2 +
                      (size_t)BM * sizeof(int) + (size_t)BM * sizeof(float);
  hipFuncSetAttribute((const void*)moe_ffn_kernel,
                      hipFuncAttributeMaxDynamicSharedMemorySize, (int)smem);
  moe_ffn_kernel<<<dim3(T_TOK / BM, NEXP), 256, smem, stream>>>(
      x, w12t, w3t, counts, idxb, wgtb, out);
}
